// TSGuard_11321533792839
// MI455X (gfx1250) — matrix operand order repaired and checked by reading, not compiled, not run on hardware
//
#include <hip/hip_runtime.h>
#include <hip/hip_bf16.h>

typedef __attribute__((ext_vector_type(16))) _Float16 v16h;
typedef __attribute__((ext_vector_type(8)))  float    v8f;

// ---------------------------------------------------------------------------
// Pass 0: zero the accumulators (workspace is poisoned 0xAA by the harness)
// ---------------------------------------------------------------------------
__global__ void k_init(unsigned* __restrict__ deg,
                       float* __restrict__ t1,
                       float* __restrict__ t2, int N) {
    int i = blockIdx.x * blockDim.x + threadIdx.x;
    if (i < N) { deg[i] = 0u; t1[i] = 0.0f; t2[i] = 0.0f; }
}

// ---------------------------------------------------------------------------
// Pass 1: in-degree (edge part; self-loop added analytically later)
// ---------------------------------------------------------------------------
__global__ void k_degree(const int* __restrict__ dst,
                         unsigned* __restrict__ deg, int E) {
    int i = blockIdx.x * blockDim.x + threadIdx.x;
    if (i < E) atomicAdd(&deg[dst[i]], 1u);
}

// ---------------------------------------------------------------------------
// Pass 2: dinv = rsqrt(deg+1) ;  y = x * dinv   (x is [N,1])
// ---------------------------------------------------------------------------
__global__ void k_prep(const float* __restrict__ x,
                       const unsigned* __restrict__ deg,
                       float* __restrict__ dinv,
                       float* __restrict__ y, int N) {
    int i = blockIdx.x * blockDim.x + threadIdx.x;
    if (i < N) {
        float d = rsqrtf((float)(deg[i] + 1u));   // self-loop => deg >= 1
        dinv[i] = d;
        y[i]    = x[i] * d;
    }
}

// ---------------------------------------------------------------------------
// Scatter-add of a per-node scalar along edges: acc[dst] += val[src]
// (used for both GCN layers; val arrays are L2-resident at 400KB)
// ---------------------------------------------------------------------------
__global__ void k_scatter(const int* __restrict__ src,
                          const int* __restrict__ dst,
                          const float* __restrict__ val,
                          float* __restrict__ acc, int E) {
    int i = blockIdx.x * blockDim.x + threadIdx.x;
    if (i < E) atomicAdd(&acc[dst[i]], val[src[i]]);
}

// ---------------------------------------------------------------------------
// Hidden transform via WMMA:
//   s1[n] = dinv[n] * (t1[n] + y[n])                     (layer-1 output scalar)
//   h2[n] = sum_k relu(s1[n]*W1[k] + b1[k]) * W2[k]      (layer-2 linear, K=64)
//   z[n]  = h2[n] * dinv[n]                              (pre-scaled message)
//
// One wave = 16 nodes. A tile: A[m,k] = relu(s1[m]*W1[k]+b1[k]) (f16), built
// per the documented 16-bit 16x32 A layout. B tile: every column = W2 chunk
// (f16), per the documented 32x16 B layout, so each D column holds the exact
// dot with W2. Two chained v_wmma_f32_16x16x32_f16 cover K=64.
// ---------------------------------------------------------------------------
__global__ __launch_bounds__(256)
void k_hidden_wmma(const float* __restrict__ t1,
                   const float* __restrict__ y,
                   const float* __restrict__ dinv,
                   const float* __restrict__ W1,
                   const float* __restrict__ b1,
                   const float* __restrict__ W2,
                   float* __restrict__ z, int N) {
    const int lane = threadIdx.x & 31;
    const int wave = threadIdx.x >> 5;
    const int base = (blockIdx.x * (blockDim.x >> 5) + wave) * 16;

    const int m    = lane & 15;
    const int node = base + m;
    float s = 0.0f;
    if (node < N) s = dinv[node] * (t1[node] + y[node]);

    const int hiA = (lane < 16) ? 0 : 8;    // A: lanes 16-31 hold K+8 group
    const int hiB = (lane < 16) ? 0 : 16;   // unused; B uses hiA, lanes 16-31 hold K+8 group

    v8f c = {};
#pragma unroll
    for (int kb = 0; kb < 64; kb += 32) {
        v16h a, b;
#pragma unroll
        for (int j = 0; j < 16; ++j) {
            // A element j -> K index per 16-bit 16x32 A layout
            int kA = kb + j + ((j < 8) ? 0 : 8) + hiA;
            float av = fmaxf(s * W1[kA] + b1[kA], 0.0f);
            a[j] = (_Float16)av;
            // B element j -> same K index as A element j; all 16 columns = W2[k]
            int kB = kb + j + ((j < 8) ? 0 : 8) + hiA;
            b[j] = (_Float16)W2[kB];
        }
        c = __builtin_amdgcn_wmma_f32_16x16x32_f16(
                /*neg_a=*/false, a, /*neg_b=*/false, b,
                /*c_mod=*/(short)0, c, /*reuse_a=*/false, /*reuse_b=*/false);
    }

    // D layout: VGPR r holds M=r (lanes 0-15) / M=8+r (lanes 16-31); all
    // columns are identical, so lane 0 extracts nodes 0..7, lane 16 nodes 8..15.
    if ((lane & 15) == 0) {
        const int off = (lane == 0) ? 0 : 8;
#pragma unroll
        for (int r = 0; r < 8; ++r) {
            int n2 = base + off + r;
            if (n2 < N) z[n2] = c[r] * dinv[n2];
        }
    }
}

// ---------------------------------------------------------------------------
// Final: out[d] = dinv[d]*(t2[d] + z[d]) + b2   (z[d] term = self-loop)
// ---------------------------------------------------------------------------
__global__ void k_final(const float* __restrict__ t2,
                        const float* __restrict__ z,
                        const float* __restrict__ dinv,
                        const float* __restrict__ b2,
                        float* __restrict__ out, int N) {
    int i = blockIdx.x * blockDim.x + threadIdx.x;
    if (i < N) out[i] = dinv[i] * (t2[i] + z[i]) + b2[0];
}

// ---------------------------------------------------------------------------
extern "C" void kernel_launch(void* const* d_in, const int* in_sizes, int n_in,
                              void* d_out, int out_size, void* d_ws, size_t ws_size,
                              hipStream_t stream) {
    const float* x  = (const float*)d_in[0];
    const int*   ei = (const int*)  d_in[1];   // [2, E] row-major (src row, dst row)
    const float* W1 = (const float*)d_in[2];   // 64
    const float* b1 = (const float*)d_in[3];   // 64
    const float* W2 = (const float*)d_in[4];   // 64
    const float* b2 = (const float*)d_in[5];   // 1
    float* out = (float*)d_out;

    const int N = in_sizes[0];
    const int E = in_sizes[1] / 2;
    const int* src = ei;
    const int* dst = ei + E;

    // Workspace layout (each array padded to 64 elements)
    size_t npad = (size_t)((N + 63) & ~63);
    float* wsf  = (float*)d_ws;
    unsigned* deg = (unsigned*)wsf;        // [N] u32
    float* dinv = wsf + 1 * npad;          // [N]
    float* y    = wsf + 2 * npad;          // [N]  x * dinv
    float* t1   = wsf + 3 * npad;          // [N]  layer-1 aggregation
    float* z    = wsf + 4 * npad;          // [N]  h2 * dinv
    float* t2   = wsf + 5 * npad;          // [N]  layer-2 aggregation

    const int T = 256;
    const int gN = (N + T - 1) / T;
    const int gE = (E + T - 1) / T;
    const int gW = (N + 127) / 128;        // 8 waves/block * 16 nodes/wave

    k_init   <<<gN, T, 0, stream>>>(deg, t1, t2, N);
    k_degree <<<gE, T, 0, stream>>>(dst, deg, E);
    k_prep   <<<gN, T, 0, stream>>>(x, deg, dinv, y, N);
    k_scatter<<<gE, T, 0, stream>>>(src, dst, y, t1, E);
    k_hidden_wmma<<<gW, T, 0, stream>>>(t1, y, dinv, W1, b1, W2, z, N);
    k_scatter<<<gE, T, 0, stream>>>(src, dst, z, t2, E);
    k_final  <<<gN, T, 0, stream>>>(t2, z, dinv, b2, out, N);
}
